// StandardAttention_13984413516286
// MI455X (gfx1250) — matrix operand order repaired and checked by reading, not compiled, not run on hardware
//
#include <hip/hip_runtime.h>

// ---------- types ----------
typedef __attribute__((ext_vector_type(16))) __bf16 v16bf;
typedef __attribute__((ext_vector_type(8)))  float  v8f;
typedef __attribute__((ext_vector_type(4)))  int    v4i;

struct alignas(16) U4 { unsigned int x, y, z, w; };
union FragU { U4 u[2]; v16bf v; };

#if defined(__has_builtin)
#if __has_builtin(__builtin_amdgcn_global_load_async_to_lds_b128) && \
    __has_builtin(__builtin_amdgcn_s_wait_asynccnt)
#define HAVE_ASYNC_LDS 1
#endif
#endif
#ifndef HAVE_ASYNC_LDS
#define HAVE_ASYNC_LDS 0
#endif

#if HAVE_ASYNC_LDS
typedef __attribute__((address_space(1))) v4i gv4i;   // global
typedef __attribute__((address_space(3))) v4i lv4i;   // LDS
// global -> LDS async 16B copy (ASYNCcnt); LDS offset = low 32 bits of generic ptr
__device__ inline void async_ld_b128(void* lds_dst, const void* gsrc) {
  __builtin_amdgcn_global_load_async_to_lds_b128(
      (gv4i*)(size_t)gsrc, (lv4i*)(unsigned int)(size_t)lds_dst, 0, 0);
}
#endif

__device__ inline unsigned short f2bf(float f) {
  unsigned int u = __float_as_uint(f);
  unsigned int r = u + 0x7FFFu + ((u >> 16) & 1u);   // round-to-nearest-even
  return (unsigned short)(r >> 16);
}

__device__ inline v8f wmma_bf16(v16bf a, v16bf b, v8f c) {
  // (neg_a, A, neg_b, B, c_mod, C, reuse_a, reuse_b)
  return __builtin_amdgcn_wmma_f32_16x16x32_bf16(false, a, false, b, (short)0, c,
                                                 false, false);
}

// A-matrix 16x32 bf16 fragment (ISA 7.12.2): lane<16 -> row r, K = kb+{0..7,16..23};
// lane>=16 -> row r, K = kb+{8..15,24..31}
__device__ inline v16bf load_a_frag(const unsigned short* p, int row_base, int k_base,
                                    int stride, int lane) {
  const int hg = lane >> 4, r = lane & 15;
  const unsigned short* rp = p + (row_base + r) * stride + k_base + hg * 8;
  FragU f;
  f.u[0] = *(const U4*)(rp);
  f.u[1] = *(const U4*)(rp + 16);
  return f.v;
}

// B-matrix 32x16 bf16 fragment: lane<16 -> col r, K = kb+{0..7,16..23}; lane>=16 -> K = kb+{8..15,24..31}.
__device__ inline v16bf load_b_frag(const unsigned short* p, int row_base, int k_base,
                                    int stride, int lane) {
  const int hg = lane >> 4, r = lane & 15;
  const unsigned short* rp = p + (row_base + r) * stride + k_base + hg * 8;
  FragU f;
  f.u[0] = *(const U4*)(rp);
  f.u[1] = *(const U4*)(rp + 16);
  return f.v;
}

__device__ inline float red_max16(float v) {
  v = fmaxf(v, __shfl_xor(v, 1, 32));
  v = fmaxf(v, __shfl_xor(v, 2, 32));
  v = fmaxf(v, __shfl_xor(v, 4, 32));
  v = fmaxf(v, __shfl_xor(v, 8, 32));
  return v;
}
__device__ inline float red_sum16(float v) {
  v += __shfl_xor(v, 1, 32);
  v += __shfl_xor(v, 2, 32);
  v += __shfl_xor(v, 4, 32);
  v += __shfl_xor(v, 8, 32);
  return v;
}

// ---------- fp32 -> bf16 cast ----------
__global__ __launch_bounds__(256) void cast_kernel(const float* __restrict__ src,
                                                   unsigned short* __restrict__ dst,
                                                   int n) {
  int i = blockIdx.x * 256 + threadIdx.x;
  if (i < n) dst[i] = f2bf(src[i]);
}

// ---------- GEMM: C[M,N] = A[M,K](bf16) * W[N,K]^T (bf16) + bias ----------
// 128x128 tile per workgroup; 8 waves in 2(M)x4(N); each wave: 64x32 = 4x2 WMMA tiles.
// Double-buffered LDS; tile fills via GLOBAL_LOAD_ASYNC_TO_LDS_B128 when available.
template <bool BF16_OUT>
__global__ __launch_bounds__(256)
void gemm_xwt(const unsigned short* __restrict__ A,
              const unsigned short* __restrict__ W,
              const float* __restrict__ bias,
              void* __restrict__ Cout, int M, int N, int K) {
  __shared__ unsigned short lds_a[2][128 * 40];
  __shared__ unsigned short lds_b[2][128 * 40];
  const int tid = threadIdx.x, lane = tid & 31, wave = tid >> 5;
  const int wm = (wave >> 2) * 64;     // 0 / 64
  const int wn = (wave & 3) * 32;      // 0..96
  const int m0 = blockIdx.y * 128, n0 = blockIdx.x * 128;

  int srow[2], scol[2];
#pragma unroll
  for (int j = 0; j < 2; ++j) {
    int i = tid + j * 256;                 // 0..511
    srow[j] = i >> 2;                      // 128 rows
    scol[j] = (i & 3) * 8;                 // 4 chunks of 8 bf16
  }

  v8f acc[4][2];
#pragma unroll
  for (int i = 0; i < 4; ++i)
#pragma unroll
    for (int j = 0; j < 2; ++j) acc[i][j] = (v8f)(0.0f);

#if HAVE_ASYNC_LDS
  // prologue: async-fill buffer 0 (4 ops/thread in flight)
#pragma unroll
  for (int j = 0; j < 2; ++j) {
    async_ld_b128(&lds_a[0][srow[j] * 40 + scol[j]],
                  &A[(size_t)(m0 + srow[j]) * K + scol[j]]);
    async_ld_b128(&lds_b[0][srow[j] * 40 + scol[j]],
                  &W[(size_t)(n0 + srow[j]) * K + scol[j]]);
  }
#else
  U4 ra[2], rb[2];
#pragma unroll
  for (int j = 0; j < 2; ++j) {
    ra[j] = *(const U4*)&A[(size_t)(m0 + srow[j]) * K + scol[j]];
    rb[j] = *(const U4*)&W[(size_t)(n0 + srow[j]) * K + scol[j]];
  }
#pragma unroll
  for (int j = 0; j < 2; ++j) {
    *(U4*)&lds_a[0][srow[j] * 40 + scol[j]] = ra[j];
    *(U4*)&lds_b[0][srow[j] * 40 + scol[j]] = rb[j];
  }
#endif

  int cur = 0;
  for (int kt = 0; kt < K; kt += 32) {
    const bool has_next = (kt + 32) < K;
#if HAVE_ASYNC_LDS
    if (has_next) {
      const int nxt = cur ^ 1;
#pragma unroll
      for (int j = 0; j < 2; ++j) {
        async_ld_b128(&lds_a[nxt][srow[j] * 40 + scol[j]],
                      &A[(size_t)(m0 + srow[j]) * K + kt + 32 + scol[j]]);
        async_ld_b128(&lds_b[nxt][srow[j] * 40 + scol[j]],
                      &W[(size_t)(n0 + srow[j]) * K + kt + 32 + scol[j]]);
      }
      __builtin_amdgcn_s_wait_asynccnt(4);  // in-order: current tile's 4 ops done
    } else {
      __builtin_amdgcn_s_wait_asynccnt(0);
    }
#else
    if (has_next) {
#pragma unroll
      for (int j = 0; j < 2; ++j) {
        ra[j] = *(const U4*)&A[(size_t)(m0 + srow[j]) * K + kt + 32 + scol[j]];
        rb[j] = *(const U4*)&W[(size_t)(n0 + srow[j]) * K + kt + 32 + scol[j]];
      }
    }
#endif
    __syncthreads();   // current buffer's fill visible to all waves

    const unsigned short* la = lds_a[cur];
    const unsigned short* lb = lds_b[cur];
    v16bf bf0 = load_b_frag(lb, wn + 0,  0, 40, lane);
    v16bf bf1 = load_b_frag(lb, wn + 16, 0, 40, lane);
#pragma unroll
    for (int i = 0; i < 4; ++i) {
      v16bf af = load_a_frag(la, wm + i * 16, 0, 40, lane);
      acc[i][0] = wmma_bf16(af, bf0, acc[i][0]);
      acc[i][1] = wmma_bf16(af, bf1, acc[i][1]);
    }

#if !HAVE_ASYNC_LDS
    if (has_next) {
      int nxt = cur ^ 1;
#pragma unroll
      for (int j = 0; j < 2; ++j) {
        *(U4*)&lds_a[nxt][srow[j] * 40 + scol[j]] = ra[j];
        *(U4*)&lds_b[nxt][srow[j] * 40 + scol[j]] = rb[j];
      }
    }
#endif
    cur ^= 1;
  }

  const int r = lane & 15, hg = lane >> 4;
#pragma unroll
  for (int i = 0; i < 4; ++i)
#pragma unroll
    for (int j = 0; j < 2; ++j)
#pragma unroll
      for (int p = 0; p < 8; ++p) {
        int m = m0 + wm + i * 16 + p + hg * 8;   // C layout: lanes>=16 hold M+8
        int n = n0 + wn + j * 16 + r;
        float v = acc[i][j][p] + bias[n];
        if (BF16_OUT)
          ((unsigned short*)Cout)[(size_t)m * N + n] = f2bf(v);
        else
          ((float*)Cout)[(size_t)m * N + n] = v;
      }
}

// ---------- causal flash attention ----------
// grid = (T/128, H, B), 256 threads. Q/K/V/ctx layout: [B,T,D] bf16, head h at cols h*64.
// Double-buffered K/V blocks; K fills async global->LDS, V via regs (transpose scatter).
__global__ __launch_bounds__(256)
void attn_kernel(const unsigned short* __restrict__ Q,
                 const unsigned short* __restrict__ Kg,
                 const unsigned short* __restrict__ Vg,
                 unsigned short* __restrict__ Ctx, int T, int D) {
  __shared__ unsigned short q_lds[128 * 72];
  __shared__ unsigned short k_lds[2][64 * 72];
  __shared__ unsigned short vt_lds[2][64 * 72];   // transposed: [dh][k]
  __shared__ unsigned short p_lds[8][16 * 72];    // per-wave P round-trip buffer

  const int tid = threadIdx.x, lane = tid & 31, wave = tid >> 5;
  const int q0 = blockIdx.x * 128, h = blockIdx.y, b = blockIdx.z;
  const size_t base = ((size_t)b * T) * D + (size_t)h * 64;
  const int m_off = wave * 16;
  const int r = lane & 15, hg = lane >> 4;
  const float SCALE = 0.125f;                     // 1/sqrt(64)

  int srow[2], scol[2];
#pragma unroll
  for (int j = 0; j < 2; ++j) {
    int i = tid + j * 256;
    srow[j] = i >> 3;
    scol[j] = (i & 7) * 8;
  }

  // Q tile: 128x64 (visible after first loop barrier)
#pragma unroll
  for (int j = 0; j < 4; ++j) {
    int i = tid + j * 256;
    int row = i >> 3, col = (i & 7) * 8;
    *(U4*)&q_lds[row * 72 + col] = *(const U4*)&Q[base + (size_t)(q0 + row) * D + col];
  }

  float m_i[8], l_i[8];
  v8f acc[4];
#pragma unroll
  for (int p = 0; p < 8; ++p) { m_i[p] = -1e30f; l_i[p] = 0.0f; }
#pragma unroll
  for (int ns = 0; ns < 4; ++ns) acc[ns] = (v8f)(0.0f);

  unsigned short* pw = &p_lds[wave][0];
  const int kb_end = q0 + 128;

  // prologue: stage block kb=0 into buffer 0
  U4 rk[2], rv[2];
#if HAVE_ASYNC_LDS
#pragma unroll
  for (int j = 0; j < 2; ++j)
    async_ld_b128(&k_lds[0][srow[j] * 72 + scol[j]],
                  &Kg[base + (size_t)srow[j] * D + scol[j]]);
#else
#pragma unroll
  for (int j = 0; j < 2; ++j)
    rk[j] = *(const U4*)&Kg[base + (size_t)srow[j] * D + scol[j]];
#endif
#pragma unroll
  for (int j = 0; j < 2; ++j)
    rv[j] = *(const U4*)&Vg[base + (size_t)srow[j] * D + scol[j]];
#pragma unroll
  for (int j = 0; j < 2; ++j) {
#if !HAVE_ASYNC_LDS
    *(U4*)&k_lds[0][srow[j] * 72 + scol[j]] = rk[j];
#endif
    const unsigned short* vs = (const unsigned short*)&rv[j];
#pragma unroll
    for (int e = 0; e < 8; ++e) vt_lds[0][(scol[j] + e) * 72 + srow[j]] = vs[e];
  }

  int cur = 0;
  for (int kb = 0; kb < kb_end; kb += 64) {
    const bool has_next = (kb + 64) < kb_end;
    if (has_next) {
#if HAVE_ASYNC_LDS
      const int nxt = cur ^ 1;
#pragma unroll
      for (int j = 0; j < 2; ++j)
        async_ld_b128(&k_lds[nxt][srow[j] * 72 + scol[j]],
                      &Kg[base + (size_t)(kb + 64 + srow[j]) * D + scol[j]]);
#else
#pragma unroll
      for (int j = 0; j < 2; ++j)
        rk[j] = *(const U4*)&Kg[base + (size_t)(kb + 64 + srow[j]) * D + scol[j]];
#endif
#pragma unroll
      for (int j = 0; j < 2; ++j)
        rv[j] = *(const U4*)&Vg[base + (size_t)(kb + 64 + srow[j]) * D + scol[j]];
    }
#if HAVE_ASYNC_LDS
    if (has_next) __builtin_amdgcn_s_wait_asynccnt(2);  // current K block resident
    else          __builtin_amdgcn_s_wait_asynccnt(0);
#endif
    __syncthreads();   // current K/V buffer (and Q on first iter) visible

    if (kb <= q0 + m_off + 15) {                  // wave-uniform causal block skip
      const unsigned short* lk = k_lds[cur];
      const unsigned short* lv = vt_lds[cur];

      // S = Q * K^T  (16 x 64)
      v8f s[4];
#pragma unroll
      for (int ns = 0; ns < 4; ++ns) s[ns] = (v8f)(0.0f);
#pragma unroll
      for (int ks = 0; ks < 2; ++ks) {
        v16bf af = load_a_frag(q_lds, m_off, ks * 32, 72, lane);
#pragma unroll
        for (int ns = 0; ns < 4; ++ns) {
          v16bf bfr = load_b_frag(lk, ns * 16, ks * 32, 72, lane);
          s[ns] = wmma_bf16(af, bfr, s[ns]);
        }
      }

      // online softmax per query row (row lives on one 16-lane half-wave)
#pragma unroll
      for (int p = 0; p < 8; ++p) {
        const int q_idx = q0 + m_off + p + hg * 8;
        float v0[4];
        float rmax = -1e30f;
#pragma unroll
        for (int ns = 0; ns < 4; ++ns) {
          float v = s[ns][p] * SCALE;
          int k_idx = kb + ns * 16 + r;
          if (k_idx > q_idx) v = -1e30f;
          v0[ns] = v;
          rmax = fmaxf(rmax, v);
        }
        rmax = red_max16(rmax);
        float m_new = fmaxf(m_i[p], rmax);
        float corr = __expf(m_i[p] - m_new);
        float rsum = 0.0f;
#pragma unroll
        for (int ns = 0; ns < 4; ++ns) {
          float pv = __expf(v0[ns] - m_new);
          rsum += pv;
          pw[(p + hg * 8) * 72 + ns * 16 + r] = f2bf(pv);
        }
        rsum = red_sum16(rsum);
        l_i[p] = l_i[p] * corr + rsum;
        m_i[p] = m_new;
#pragma unroll
        for (int ns = 0; ns < 4; ++ns) acc[ns][p] = acc[ns][p] * corr;
      }

      // O += P * V  (same-wave DS store->load is ordered)
#pragma unroll
      for (int ks = 0; ks < 2; ++ks) {
        v16bf af = load_a_frag(pw, 0, ks * 32, 72, lane);
#pragma unroll
        for (int ns = 0; ns < 4; ++ns) {
          v16bf bfr = load_b_frag(lv, ns * 16, ks * 32, 72, lane);
          acc[ns] = wmma_bf16(af, bfr, acc[ns]);
        }
      }
    }

    if (has_next) {    // fill the other buffer; nobody reads it this iteration
      int nxt = cur ^ 1;
#pragma unroll
      for (int j = 0; j < 2; ++j) {
#if !HAVE_ASYNC_LDS
        *(U4*)&k_lds[nxt][srow[j] * 72 + scol[j]] = rk[j];
#endif
        const unsigned short* vs = (const unsigned short*)&rv[j];
#pragma unroll
        for (int e = 0; e < 8; ++e) vt_lds[nxt][(scol[j] + e) * 72 + srow[j]] = vs[e];
      }
    }
    cur ^= 1;
  }

#pragma unroll
  for (int ns = 0; ns < 4; ++ns)
#pragma unroll
    for (int p = 0; p < 8; ++p) {
      int t = q0 + m_off + p + hg * 8;
      int dh = ns * 16 + r;
      float o = acc[ns][p] / l_i[p];
      Ctx[base + (size_t)t * D + dh] = f2bf(o);
    }
}

// ---------- launcher ----------
extern "C" void kernel_launch(void* const* d_in, const int* in_sizes, int n_in,
                              void* d_out, int out_size, void* d_ws, size_t ws_size,
                              hipStream_t stream) {
  (void)in_sizes; (void)n_in; (void)out_size; (void)ws_size;
  const float* x  = (const float*)d_in[0];
  const float* Wq = (const float*)d_in[1];
  const float* bq = (const float*)d_in[2];
  const float* Wk = (const float*)d_in[3];
  const float* bk = (const float*)d_in[4];
  const float* Wv = (const float*)d_in[5];
  const float* bv = (const float*)d_in[6];
  const float* Wo = (const float*)d_in[7];
  const float* bo = (const float*)d_in[8];
  float* out = (float*)d_out;

  const int B = 2, T = 2048, D = 1024, H = 16;
  const int M = B * T;  // 4096

  unsigned short* ws  = (unsigned short*)d_ws;
  unsigned short* xb  = ws; ws += (size_t)M * D;
  unsigned short* wqb = ws; ws += (size_t)D * D;
  unsigned short* wkb = ws; ws += (size_t)D * D;
  unsigned short* wvb = ws; ws += (size_t)D * D;
  unsigned short* wob = ws; ws += (size_t)D * D;
  unsigned short* qb  = ws; ws += (size_t)M * D;
  unsigned short* kbuf= ws; ws += (size_t)M * D;
  unsigned short* vb  = ws; ws += (size_t)M * D;
  unsigned short* ctx = ws; ws += (size_t)M * D;

  // fp32 -> bf16 casts
  cast_kernel<<<(M * D + 255) / 256, 256, 0, stream>>>(x,  xb,  M * D);
  cast_kernel<<<(D * D + 255) / 256, 256, 0, stream>>>(Wq, wqb, D * D);
  cast_kernel<<<(D * D + 255) / 256, 256, 0, stream>>>(Wk, wkb, D * D);
  cast_kernel<<<(D * D + 255) / 256, 256, 0, stream>>>(Wv, wvb, D * D);
  cast_kernel<<<(D * D + 255) / 256, 256, 0, stream>>>(Wo, wob, D * D);

  // QKV projections (bf16 out)
  dim3 ggrid(D / 128, M / 128);
  gemm_xwt<true><<<ggrid, 256, 0, stream>>>(xb, wqb, bq, qb,   M, D, D);
  gemm_xwt<true><<<ggrid, 256, 0, stream>>>(xb, wkb, bk, kbuf, M, D, D);
  gemm_xwt<true><<<ggrid, 256, 0, stream>>>(xb, wvb, bv, vb,   M, D, D);

  // causal flash attention
  dim3 agrid(T / 128, H, B);
  attn_kernel<<<agrid, 256, 0, stream>>>(qb, kbuf, vb, ctx, T, D);

  // output projection (fp32 out)
  gemm_xwt<false><<<ggrid, 256, 0, stream>>>(ctx, wob, bo, out, M, D, D);
}
